// VCM_Encoder_23321672417649
// MI455X (gfx1250) — matrix operand order repaired and checked by reading, not compiled, not run on hardware
//
#include <hip/hip_runtime.h>
#include <hip/hip_bf16.h>

// ---------------- problem constants ----------------
#define BB   32
#define TT   256
#define RR   10000
#define KK   1000
#define KK2  100
#define KP   1024        // K padded for WMMA tiling
#define N2P  112         // K2 padded to 7x16
#define MM   (BB*TT)     // 8192 rows
#define NSORT 16384      // power-of-2 pad of RR
static const size_t BTR = (size_t)BB * TT * RR;   // 81,920,000

typedef __bf16 bf16;
typedef __bf16 v16bf __attribute__((ext_vector_type(16)));
typedef __bf16 v8bf  __attribute__((ext_vector_type(8)));
typedef float  v8f   __attribute__((ext_vector_type(8)));
typedef unsigned int v4u __attribute__((ext_vector_type(4)));
typedef int          v8i __attribute__((ext_vector_type(8)));
typedef int          v4i __attribute__((ext_vector_type(4)));

static __device__ inline bf16 f2bf(float f) {
  unsigned u = __float_as_uint(f);
  u += 0x7FFFu + ((u >> 16) & 1u);          // round-to-nearest-even
  unsigned short h = (unsigned short)(u >> 16);
  bf16 b;
  __builtin_memcpy(&b, &h, 2);
  return b;
}

// LDS byte offset of a __shared__ object (flat-address low 32 bits == LDS offset)
static __device__ inline unsigned lds_off(const void* p) {
  return (unsigned)(size_t)p;
}

// ---- TDM: DMA a 2D bf16 tile (tileW x tileH elems, row stride strideElems)
//      from global memory into LDS at ldsAddr. Wave-level op, EXEC-ignored,
//      tracked by TENSORcnt. D# layout per CDNA5 ISA ch.8.
//      This toolchain exposes the 6-arg builtin:
//      (v4u g0, v8i g1, v4i g2, v4i g3, v8i pad, i32 cpol). ----
static __device__ inline void tdm_load_bf16_2d(unsigned ldsAddr, const void* gptr,
                                               unsigned tileW, unsigned tileH,
                                               unsigned tensorW, unsigned tensorH,
                                               unsigned long long strideElems) {
  unsigned long long ga = (unsigned long long)(size_t)gptr;
  v4u g0;
  g0[0] = 1u;                                           // count=1, user descriptor
  g0[1] = ldsAddr;                                      // lds_addr (bytes)
  g0[2] = (unsigned)(ga & 0xFFFFFFFFull);               // global_addr[31:0]
  g0[3] = (unsigned)((ga >> 32) & 0x01FFFFFFull)        // global_addr[56:32]
        | (2u << 30);                                   // type=2 ("image")
  v8i g1;
  g1[0] = (int)(1u << 16);                              // data_size=1 -> 2 bytes
  g1[1] = (int)((tensorW & 0xFFFFu) << 16);             // tensor_dim0[15:0]
  g1[2] = (int)((tensorW >> 16) | ((tensorH & 0xFFFFu) << 16));
  g1[3] = (int)((tensorH >> 16) | (tileW << 16));       // tile_dim0
  g1[4] = (int)tileH;                                   // tile_dim1 (tile_dim2=0)
  g1[5] = (int)(unsigned)(strideElems & 0xFFFFFFFFull); // tensor_dim0_stride[31:0]
  g1[6] = (int)(unsigned)((strideElems >> 32) & 0xFFFFull);
  g1[7] = 0;                                            // tensor_dim1_stride=0 (2D)
  v4i z4 = {0, 0, 0, 0};
  v8i z8 = {0, 0, 0, 0, 0, 0, 0, 0};
  __builtin_amdgcn_tensor_load_to_lds(g0, g1, z4, z4, z8, 0);
}

// ---- 1. per-(b,r): NaN-zero, sum of squares over T, dot with wp -> invnorm, score ----
__global__ void __launch_bounds__(256) k_norm_score(const float* __restrict__ x,
                                                    const float* __restrict__ wp,
                                                    const float* __restrict__ bp,
                                                    float* __restrict__ invnorm,
                                                    float* __restrict__ score) {
  __shared__ float swp[TT];
  int tid = threadIdx.x;
  swp[tid] = wp[tid];
  __syncthreads();
  int r = blockIdx.x * 256 + tid;
  int b = blockIdx.y;
  if (r >= RR) return;
  const float* xp = x + (size_t)b * TT * RR + r;
  float ss = 0.f, dot = 0.f;
  int anynan = 0;
  #pragma unroll 4
  for (int t = 0; t < TT; ++t) {
    float v = xp[(size_t)t * RR];
    int isn = (v != v);
    float v0 = isn ? 0.f : v;
    ss += v0 * v0;
    dot += v0 * swp[t];
    anynan |= isn;
  }
  float inv = 1.f / fmaxf(sqrtf(ss), 1e-12f);
  invnorm[b * RR + r] = inv;
  score[b * RR + r] = anynan ? -10000.f : (dot * inv + bp[0]);
}

// ---- 2a. pack scores into sortable u64 keys: score desc, index asc ----
__global__ void __launch_bounds__(256) k_pack(const float* __restrict__ score,
                                              unsigned long long* __restrict__ buf) {
  int g = blockIdx.x * 256 + threadIdx.x;     // over BB*NSORT
  int b = g >> 14, s = g & (NSORT - 1);
  unsigned long long p;
  if (s < RR) {
    unsigned u = __float_as_uint(score[b * RR + s]);
    unsigned m = (u & 0x80000000u) ? ~u : (u | 0x80000000u);  // monotone asc map
    p = ((unsigned long long)(~m) << 32) | (unsigned)s;       // asc sort == score desc
  } else {
    p = ~0ull;                                                // pad sorts last
  }
  buf[g] = p;
}

// ---- 2b. one bitonic compare-exchange pass (ascending) over each batch segment ----
__global__ void __launch_bounds__(256) k_bitonic(unsigned long long* __restrict__ buf,
                                                 int ksz, int j) {
  int g = blockIdx.x * 256 + threadIdx.x;
  int b = g >> 14, i = g & (NSORT - 1);
  int ij = i ^ j;
  if (ij <= i) return;
  unsigned long long* p = buf + ((size_t)b << 14);
  unsigned long long a = p[i], c = p[ij];
  bool up = ((i & ksz) == 0);
  if (up ? (a > c) : (a < c)) { p[i] = c; p[ij] = a; }
}

// ---- 2c. extract top-K indices ----
__global__ void __launch_bounds__(256) k_extract(const unsigned long long* __restrict__ buf,
                                                 int* __restrict__ idxw,
                                                 float* __restrict__ idxf) {
  int g = blockIdx.x * 256 + threadIdx.x;     // BB*KK
  if (g >= BB * KK) return;
  int b = g / KK, kk = g - b * KK;
  int idx = (int)(unsigned)(buf[((size_t)b << 14) + kk] & 0xFFFFFFFFull);
  idxw[g] = idx;
  idxf[g] = (float)idx;
}

__global__ void __launch_bounds__(256) k_zero_maskc(float* __restrict__ mc) {
  int g = blockIdx.x * 256 + threadIdx.x;
  if (g < BB * RR) mc[g] = 0.f;
}

__global__ void __launch_bounds__(256) k_scatter(const int* __restrict__ idxw,
                                                 float* __restrict__ mc) {
  int g = blockIdx.x * 256 + threadIdx.x;     // BB*KK
  if (g >= BB * KK) return;
  int b = g / KK;
  mc[b * RR + idxw[g]] = 1.f;
}

// ---- 3. big elementwise outputs: mask (broadcast over T) and border_mask ----
//      Non-temporal: 1 GB single-pass stream; keep it out of L2 so the GEMM
//      working set and gather re-reads of x stay resident. ----
__global__ void __launch_bounds__(256) k_outputs(const float* __restrict__ x,
                                                 const float* __restrict__ mc,
                                                 float* __restrict__ maskO,
                                                 float* __restrict__ borderO) {
  size_t i = (size_t)blockIdx.x * 256 + threadIdx.x;
  if (i >= BTR) return;
  __builtin_prefetch(&x[i + 65536], 0, 1);     // speculative stream-ahead
  int r = (int)(i % RR);
  int b = (int)((i / RR) / TT);
  float v = __builtin_nontemporal_load(&x[i]);
  __builtin_nontemporal_store((v != v) ? 1.f : 0.f, &borderO[i]);
  __builtin_nontemporal_store(mc[b * RR + r], &maskO[i]);
}

// ---- 4. gather selected columns, apply normalization, emit bf16 (K padded to 1024) ----
__global__ void __launch_bounds__(256) k_gather(const float* __restrict__ x,
                                                const int* __restrict__ idxw,
                                                const float* __restrict__ invnorm,
                                                bf16* __restrict__ xg) {
  int kk = blockIdx.x * 256 + threadIdx.x;    // 0..1023
  int bt = blockIdx.y;                        // 0..8191
  int b  = bt >> 8;
  float v = 0.f;
  if (kk < KK) {
    int r = idxw[b * KK + kk];
    float xv = x[(size_t)bt * RR + r];
    if (xv != xv) xv = 0.f;
    v = xv * invnorm[b * RR + r];
  }
  xg[(size_t)bt * KP + kk] = f2bf(v);
}

// ---- weight conversion to padded bf16 ----
__global__ void __launch_bounds__(256) k_cvt_w1(const float* __restrict__ W1, bf16* __restrict__ W1b) {
  int k = blockIdx.x * 256 + threadIdx.x;     // 0..1023
  int j = blockIdx.y;                         // 0..1023
  float v = (j < KK && k < KK) ? W1[j * KK + k] : 0.f;
  W1b[(size_t)j * KP + k] = f2bf(v);
}

__global__ void __launch_bounds__(256) k_cvt_w2(const float* __restrict__ W2, bf16* __restrict__ W2b) {
  int k = blockIdx.x * 256 + threadIdx.x;     // 0..1023
  int j = blockIdx.y;                         // 0..111
  float v = (j < KK2 && k < KK) ? W2[j * KK + k] : 0.f;
  W2b[(size_t)j * KP + k] = f2bf(v);
}

// ---- 5. GEMM1: H[8192,1024] = xg[8192,1024] * W1b[1024,1024]^T + b1 ----
// bf16 WMMA, f32 acc. TDM double-buffered tile staging: wave 0 DMAs the next
// k-step's A/B tiles into the alternate LDS buffer while all 8 waves compute,
// then s_wait_tensorcnt + workgroup barrier publishes the buffer.
__global__ void __launch_bounds__(256) k_gemm1(const bf16* __restrict__ A,
                                               const bf16* __restrict__ Bw,
                                               const float* __restrict__ b1,
                                               float* __restrict__ H) {
  __shared__ bf16 As[2][128 * 32];
  __shared__ bf16 Bs[2][64 * 32];
  int tid  = threadIdx.x;
  int w    = tid >> 5;
  int lane = tid & 31;
  int m0 = blockIdx.y * 128;
  int n0 = blockIdx.x * 64;
  int mL = lane & 15;                 // row/col within 16
  int hb = lane >> 4;                 // lane half
  int abase = hb * 8;                 // A-frag K-octet select (ISA 16-bit A layout)

  v8f acc[4];
  #pragma unroll
  for (int nt = 0; nt < 4; ++nt)
    #pragma unroll
    for (int v = 0; v < 8; ++v) acc[nt][v] = 0.f;

  unsigned ldsA0 = lds_off(&As[0][0]), ldsA1 = lds_off(&As[1][0]);
  unsigned ldsB0 = lds_off(&Bs[0][0]), ldsB1 = lds_off(&Bs[1][0]);

  if (w == 0) {  // prologue: DMA tile k0=0 into buffer 0
    tdm_load_bf16_2d(ldsA0, A  + (size_t)m0 * KP, 32, 128, KP, MM,  KP);
    tdm_load_bf16_2d(ldsB0, Bw + (size_t)n0 * KP, 32, 64,  KP, KP,  KP);
  }

  int cur = 0;
  for (int k0 = 0; k0 < KP; k0 += 32) {
    if (w == 0) {
      if (k0 + 32 < KP) {  // DMA next tile into the other buffer, keep it in flight
        unsigned la = cur ? ldsA0 : ldsA1;
        unsigned lb = cur ? ldsB0 : ldsB1;
        tdm_load_bf16_2d(la, A  + (size_t)m0 * KP + (k0 + 32), 32, 128, KP, MM, KP);
        tdm_load_bf16_2d(lb, Bw + (size_t)n0 * KP + (k0 + 32), 32, 64,  KP, KP, KP);
        __builtin_amdgcn_s_wait_tensorcnt(2);   // current pair done; next pair in flight
      } else {
        __builtin_amdgcn_s_wait_tensorcnt(0);   // last pair done
      }
    }
    __syncthreads();

    // A fragment: lane holds row mL, K-octets [abase..abase+7] and [16+abase..]
    const bf16* Arow = &As[cur][(w * 16 + mL) * 32];
    v8bf alo = *(const v8bf*)(Arow + abase);
    v8bf ahi = *(const v8bf*)(Arow + 16 + abase);
    v16bf af;
    #pragma unroll
    for (int i = 0; i < 8; ++i) { af[i] = alo[i]; af[8 + i] = ahi[i]; }

    #pragma unroll
    for (int nt = 0; nt < 4; ++nt) {
      // B fragment: lane holds col n = nt*16+mL, K-octets [hb*8..hb*8+7] and [16+hb*8..]
      const bf16* Bcol = &Bs[cur][(nt * 16 + mL) * 32];
      v8bf blo = *(const v8bf*)(Bcol + abase);
      v8bf bhi = *(const v8bf*)(Bcol + 16 + abase);
      v16bf bfv;
      #pragma unroll
      for (int i = 0; i < 8; ++i) { bfv[i] = blo[i]; bfv[8 + i] = bhi[i]; }
      acc[nt] = __builtin_amdgcn_wmma_f32_16x16x32_bf16(
          false, af, false, bfv, (short)0, acc[nt], false, false);
    }
    __syncthreads();
    cur ^= 1;
  }

  #pragma unroll
  for (int nt = 0; nt < 4; ++nt) {
    int ng = n0 + nt * 16 + mL;
    float bias = (ng < KK) ? b1[ng] : 0.f;
    #pragma unroll
    for (int v = 0; v < 8; ++v) {
      int mg = m0 + w * 16 + v + hb * 8;     // C layout: M = v + 8*laneHalf
      H[(size_t)mg * KP + ng] = acc[nt][v] + bias;
    }
  }
}

// ---- 6. BatchNorm1d(T): stats over (B,K) per channel t; emit bf16 (cols>=1000 zeroed) ----
__global__ void __launch_bounds__(256) k_bn(const float* __restrict__ H,
                                            const float* __restrict__ gamma,
                                            const float* __restrict__ beta,
                                            bf16* __restrict__ Hb) {
  int t = blockIdx.x, tid = threadIdx.x;
  float s = 0.f, ss = 0.f;
  for (int b = 0; b < BB; ++b) {
    const float* row = H + (size_t)(b * TT + t) * KP;
    for (int k = tid; k < KK; k += 256) { float v = row[k]; s += v; ss += v * v; }
  }
  __shared__ float rs[256], rss[256];
  rs[tid] = s; rss[tid] = ss;
  __syncthreads();
  for (int o = 128; o > 0; o >>= 1) {
    if (tid < o) { rs[tid] += rs[tid + o]; rss[tid] += rss[tid + o]; }
    __syncthreads();
  }
  float mu  = rs[0] * (1.f / (BB * KK));
  float var = rss[0] * (1.f / (BB * KK)) - mu * mu;
  float sc  = rsqrtf(var + 1e-5f) * gamma[t];
  float sh  = beta[t];
  for (int b = 0; b < BB; ++b) {
    const float* row = H + (size_t)(b * TT + t) * KP;
    bf16* orow = Hb + (size_t)(b * TT + t) * KP;
    for (int k = tid; k < KP; k += 256)
      orow[k] = f2bf(k < KK ? (row[k] - mu) * sc + sh : 0.f);
  }
}

// ---- 7. GEMM2: out[8192,100] = Hb[8192,1024] * W2b[112,1024]^T + b2 ----
// TDM double-buffered like GEMM1; wave w -> 7 n-subtiles of the 128x112 block tile.
__global__ void __launch_bounds__(256) k_gemm2(const bf16* __restrict__ A,
                                               const bf16* __restrict__ Bw,
                                               const float* __restrict__ b2,
                                               float* __restrict__ out) {
  __shared__ bf16 As[2][128 * 32];
  __shared__ bf16 Bs[2][N2P * 32];
  int tid  = threadIdx.x;
  int w    = tid >> 5;
  int lane = tid & 31;
  int m0 = blockIdx.x * 128;
  int mL = lane & 15;
  int hb = lane >> 4;
  int abase = hb * 8;

  v8f acc[7];
  #pragma unroll
  for (int nt = 0; nt < 7; ++nt)
    #pragma unroll
    for (int v = 0; v < 8; ++v) acc[nt][v] = 0.f;

  unsigned ldsA0 = lds_off(&As[0][0]), ldsA1 = lds_off(&As[1][0]);
  unsigned ldsB0 = lds_off(&Bs[0][0]), ldsB1 = lds_off(&Bs[1][0]);

  if (w == 0) {
    tdm_load_bf16_2d(ldsA0, A + (size_t)m0 * KP, 32, 128, KP, MM,  KP);
    tdm_load_bf16_2d(ldsB0, Bw,                  32, N2P, KP, N2P, KP);
  }

  int cur = 0;
  for (int k0 = 0; k0 < KP; k0 += 32) {
    if (w == 0) {
      if (k0 + 32 < KP) {
        unsigned la = cur ? ldsA0 : ldsA1;
        unsigned lb = cur ? ldsB0 : ldsB1;
        tdm_load_bf16_2d(la, A  + (size_t)m0 * KP + (k0 + 32), 32, 128, KP, MM,  KP);
        tdm_load_bf16_2d(lb, Bw + (k0 + 32),                   32, N2P, KP, N2P, KP);
        __builtin_amdgcn_s_wait_tensorcnt(2);
      } else {
        __builtin_amdgcn_s_wait_tensorcnt(0);
      }
    }
    __syncthreads();

    const bf16* Arow = &As[cur][(w * 16 + mL) * 32];
    v8bf alo = *(const v8bf*)(Arow + abase);
    v8bf ahi = *(const v8bf*)(Arow + 16 + abase);
    v16bf af;
    #pragma unroll
    for (int i = 0; i < 8; ++i) { af[i] = alo[i]; af[8 + i] = ahi[i]; }

    #pragma unroll
    for (int nt = 0; nt < 7; ++nt) {
      const bf16* Bcol = &Bs[cur][(nt * 16 + mL) * 32];
      v8bf blo = *(const v8bf*)(Bcol + abase);
      v8bf bhi = *(const v8bf*)(Bcol + 16 + abase);
      v16bf bfv;
      #pragma unroll
      for (int i = 0; i < 8; ++i) { bfv[i] = blo[i]; bfv[8 + i] = bhi[i]; }
      acc[nt] = __builtin_amdgcn_wmma_f32_16x16x32_bf16(
          false, af, false, bfv, (short)0, acc[nt], false, false);
    }
    __syncthreads();
    cur ^= 1;
  }

  #pragma unroll
  for (int nt = 0; nt < 7; ++nt) {
    int ng = nt * 16 + mL;
    if (ng >= KK2) continue;
    float bias = b2[ng];
    #pragma unroll
    for (int v = 0; v < 8; ++v) {
      int mg = m0 + w * 16 + v + hb * 8;
      out[(size_t)mg * KK2 + ng] = acc[nt][v] + bias;
    }
  }
}

// ---------------- host-side launch ----------------
extern "C" void kernel_launch(void* const* d_in, const int* in_sizes, int n_in,
                              void* d_out, int out_size, void* d_ws, size_t ws_size,
                              hipStream_t stream) {
  const float* x     = (const float*)d_in[0];
  const float* wp    = (const float*)d_in[1];
  const float* bp    = (const float*)d_in[2];
  const float* W1    = (const float*)d_in[3];
  const float* b1    = (const float*)d_in[4];
  const float* gamma = (const float*)d_in[5];
  const float* beta  = (const float*)d_in[6];
  const float* W2    = (const float*)d_in[7];
  const float* b2    = (const float*)d_in[8];

  // output tuple layout (all float): out | mask | border_mask | idx
  float* outp    = (float*)d_out;
  float* maskO   = outp + (size_t)MM * KK2;
  float* borderO = maskO + BTR;
  float* idxf    = borderO + BTR;

  // workspace carve-up (all sizes 256B-aligned)
  char* wsb = (char*)d_ws;
  float* invnorm = (float*)wsb;                       wsb += (size_t)BB * RR * 4;
  float* score   = (float*)wsb;                       wsb += (size_t)BB * RR * 4;
  unsigned long long* sortbuf = (unsigned long long*)wsb; wsb += (size_t)BB * NSORT * 8;
  int*   idxw    = (int*)wsb;                         wsb += (size_t)BB * KK * 4;
  float* maskc   = (float*)wsb;                       wsb += (size_t)BB * RR * 4;
  bf16*  xg      = (bf16*)wsb;                        wsb += (size_t)MM * KP * 2;
  bf16*  W1b     = (bf16*)wsb;                        wsb += (size_t)KP * KP * 2;
  bf16*  W2b     = (bf16*)wsb;                        wsb += (size_t)N2P * KP * 2;
  float* H       = (float*)wsb;                       wsb += (size_t)MM * KP * 4;
  bf16*  Hb      = (bf16*)wsb;                        wsb += (size_t)MM * KP * 2;

  // 1. normalize stats + projection score
  k_norm_score<<<dim3((RR + 255) / 256, BB), 256, 0, stream>>>(x, wp, bp, invnorm, score);

  // 2. top-K via packed-u64 bitonic sort (ascending == score desc, idx asc)
  k_pack<<<(BB * NSORT) / 256, 256, 0, stream>>>(score, sortbuf);
  for (int ksz = 2; ksz <= NSORT; ksz <<= 1)
    for (int j = ksz >> 1; j > 0; j >>= 1)
      k_bitonic<<<(BB * NSORT) / 256, 256, 0, stream>>>(sortbuf, ksz, j);
  k_extract<<<(BB * KK) / 256, 256, 0, stream>>>(sortbuf, idxw, idxf);
  k_zero_maskc<<<(BB * RR + 255) / 256, 256, 0, stream>>>(maskc);
  k_scatter<<<(BB * KK) / 256, 256, 0, stream>>>(idxw, maskc);

  // 3. big elementwise outputs (roofline-dominant single pass, NT policy)
  k_outputs<<<(unsigned)((BTR + 255) / 256), 256, 0, stream>>>(x, maskc, maskO, borderO);

  // 4. gather + bf16 conversion
  k_gather<<<dim3(KP / 256, MM), 256, 0, stream>>>(x, idxw, invnorm, xg);
  k_cvt_w1<<<dim3(KP / 256, KP), 256, 0, stream>>>(W1, W1b);
  k_cvt_w2<<<dim3(KP / 256, N2P), 256, 0, stream>>>(W2, W2b);

  // 5. GEMM1 (bf16 WMMA + TDM staging) -> H
  k_gemm1<<<dim3(KP / 64, MM / 128), 256, 0, stream>>>(xg, W1b, b1, H);

  // 6. BatchNorm per channel t -> Hb (bf16)
  k_bn<<<TT, 256, 0, stream>>>(H, gamma, beta, Hb);

  // 7. GEMM2 (bf16 WMMA + TDM staging) -> out
  k_gemm2<<<MM / 128, 256, 0, stream>>>(Hb, W2b, b2, outp);
}
